// FFT3D_6820408066157
// MI455X (gfx1250) — matrix operand order repaired and checked by reading, not compiled, not run on hardware
//
#include <hip/hip_runtime.h>

typedef __attribute__((ext_vector_type(16))) _Float16 v16h;
typedef __attribute__((ext_vector_type(8)))  float    v8f;

#define DB 2
#define DD 16
#define DH 64
#define DW 64
#define DC 256
#define WF 33
#define NB 8
#define BLK 32
#define LAMBDA_ 0.01f

#define ROWB 272                 // padded LDS pane row stride in bytes (17*16, keeps B128 align)
#define NROW (DD * DH)           // 1024 pane rows of 32 complex channels (256B payload each)

// ---------------- compile-time twiddle tables -----------------------------
constexpr double KPI = 3.14159265358979323846264338327950288;

constexpr double tpoly_sin(double x) {
  double t = x, s = x;
  for (int n = 1; n < 14; ++n) { t *= -(x * x) / double((2 * n) * (2 * n + 1)); s += t; }
  return s;
}
constexpr double tpoly_cos(double x) {
  double t = 1.0, s = 1.0;
  for (int n = 1; n < 14; ++n) { t *= -(x * x) / double((2 * n - 1) * (2 * n)); s += t; }
  return s;
}
struct Tw64 {
  float c[32]; float s[32];
  constexpr Tw64() : c(), s() {
    for (int k = 0; k < 32; ++k) {
      double a = 2.0 * KPI * (double)k / 64.0;
      c[k] = (float)tpoly_cos(a);
      s[k] = (float)tpoly_sin(a);
    }
  }
};
constexpr Tw64 TW{};

constexpr int brev_c(int i, int ln) {
  int r = 0;
  for (int b = 0; b < ln; ++b) r |= ((i >> b) & 1) << (ln - 1 - b);
  return r;
}

// In-register radix-2 DIT FFT, fully unrolled, twiddles folded to literals.
template <int N, int LOGN, int SIGN>
__device__ __forceinline__ void fft_reg(float* re, float* im) {
#pragma unroll
  for (int i = 0; i < N; ++i) {
    const int j = brev_c(i, LOGN);
    if (j > i) {
      float t = re[i]; re[i] = re[j]; re[j] = t;
      t = im[i]; im[i] = im[j]; im[j] = t;
    }
  }
#pragma unroll
  for (int s = 1; s <= LOGN; ++s) {
    const int m  = 1 << s;
    const int mh = m >> 1;
#pragma unroll
    for (int k = 0; k < mh; ++k) {
      const int   tj = k * (64 / m);   // W_m^k == W_64^(k*64/m)
      const float wr = TW.c[tj];
      const float wi = (SIGN < 0) ? -TW.s[tj] : TW.s[tj];
#pragma unroll
      for (int i0 = k; i0 < N; i0 += m) {
        const int   i1 = i0 + mh;
        const float tr = wr * re[i1] - wi * im[i1];
        const float ti = wr * im[i1] + wi * re[i1];
        re[i1] = re[i0] - tr; im[i1] = im[i0] - ti;
        re[i0] += tr;         im[i0] += ti;
      }
    }
  }
}

// ---------------- async global<->LDS copy (CDNA5, ASYNCcnt) ---------------
__device__ __forceinline__ void async_g2l_b128(unsigned lds_off, const void* gaddr) {
  asm volatile("global_load_async_to_lds_b128 %0, %1, off"
               :: "v"(lds_off), "v"((unsigned long long)(uintptr_t)gaddr)
               : "memory");
}
__device__ __forceinline__ void async_l2g_b128(void* gaddr, unsigned lds_off) {
  asm volatile("global_store_async_from_lds_b128 %0, %1, off"
               :: "v"((unsigned long long)(uintptr_t)gaddr), "v"(lds_off)
               : "memory");
}
__device__ __forceinline__ void wait_async0() {
  asm volatile("s_wait_asynccnt 0x0" ::: "memory");
}

// ---------------- K0: pack weights into WMMA B-operand layout (f16) -------
// dest index: [layer][part][blk][ntile][lane][e(=16 halves)]
// B (KxN, 16-bit): lane -> n = lane&15, lh = lane>>4 ; VGPR i half j -> K = (i<4?0:16) + 8*lh + 2*(i&3) + j
__global__ __launch_bounds__(256) void k_pack_w(const float* __restrict__ w1,
                                                const float* __restrict__ w2,
                                                _Float16* __restrict__ wpk) {
  int idx = blockIdx.x * blockDim.x + threadIdx.x;
  if (idx >= 2 * 2 * NB * 2 * 32 * 16) return;
  const int e     = idx & 15;
  const int lane  = (idx >> 4) & 31;
  const int t     = (idx >> 9) & 1;
  const int blk   = (idx >> 10) & 7;
  const int part  = (idx >> 13) & 1;
  const int layer = (idx >> 14) & 1;
  const int i = e >> 1, j = e & 1;
  const int lh = lane >> 4, n = lane & 15;
  const int k = ((i < 4) ? 0 : 16) + 8 * lh + 2 * (i & 3) + j;
  const int o = t * 16 + n;
  const float* w = layer ? w2 : w1;
  wpk[idx] = (_Float16)w[(((size_t)part * NB + blk) * BLK + k) * BLK + o];
}

// ---------------- K1: real FFT along W (ortho scale folded) ---------------
__global__ __launch_bounds__(256) void k_fft_w_fwd(const float* __restrict__ x,
                                                   float2* __restrict__ Xf) {
  const int line = blockIdx.x;        // (b*DD + d)*DH + h
  const int c    = threadIdx.x;
  const int h    = line % DH;
  const int bd   = line / DH;
  const float* src = x + (((size_t)bd * DH + h) * DW) * DC + c;
  float re[DW], im[DW];
#pragma unroll
  for (int w = 0; w < DW; ++w) { re[w] = src[(size_t)w * DC]; im[w] = 0.f; }
  fft_reg<64, 6, -1>(re, im);
  float2* dst = Xf + (((size_t)bd * DH + h) * WF) * DC + c;
  const float sc = 1.0f / 256.0f;     // ortho norm for the whole forward transform
#pragma unroll
  for (int wf = 0; wf < WF; ++wf) dst[(size_t)wf * DC] = make_float2(re[wf] * sc, im[wf] * sc);
}

// ---------------- K2: fused pane kernel -----------------------------------
// One workgroup = one (b, wf, channel-block): 16x64 points x 32 complex channels
// staged in 312KB of LDS (CDNA5 WGP has 320KB). Does H-FFT, D-FFT, the complex
// block-MLP on WMMA, inverse D-FFT, inverse H-FFT, all in LDS; global traffic
// is one async read + one async write of the pane.
__global__ __launch_bounds__(256) void k_pane(float2* __restrict__ Xf,
                                              const _Float16* __restrict__ wpk,
                                              const float* __restrict__ b1,
                                              const float* __restrict__ b2) {
  __shared__ __align__(16) unsigned char pane[ROWB * NROW];   // 278,528 B
  __shared__ float hidr[NB][16][33];                          // padded: conflict-free gather
  __shared__ float hidi[NB][16][33];                          // 2 * 16,896 B

  const int grp  = blockIdx.x;            // (b*WF + wf)*NB + blk
  const int blk  = grp % NB;
  const int wf   = (grp / NB) % WF;
  const int b    = grp / (NB * WF);
  const int t    = threadIdx.x;
  const int wave = t >> 5;
  const int lane = t & 31;
  const int cb   = blk * BLK;

  // pane row r = d*DH + h ; global float2 element (r, cc) at gsrc + r*WF*DC + cc
  float2* gpane = Xf + ((((size_t)b * DD) * DH) * WF + wf) * DC + cb;

  // ---- stage pane: 1024 rows x 256B via async global->LDS (ASYNCcnt) ----
  for (int it = 0; it < 64; ++it) {
    const int q   = it * 256 + t;
    const int row = q >> 4;
    const int ch  = q & 15;               // 16B chunk within row
    const char* ga = (const char*)(gpane + (size_t)row * (WF * DC)) + ch * 16;
    async_g2l_b128((unsigned)(uintptr_t)(pane + row * ROWB + ch * 16), ga);
  }
  wait_async0();
  __syncthreads();

  float re[DH], im[DH];

  // ---- forward H-FFT: wave <-> d, lanes <-> channel (bank-conflict-free) ----
  for (int rep = 0; rep < 2; ++rep) {
    const int d = wave + 8 * rep;
#pragma unroll
    for (int h = 0; h < DH; ++h) {
      const float2 v = *(const float2*)(pane + (d * DH + h) * ROWB + lane * 8);
      re[h] = v.x; im[h] = v.y;
    }
    fft_reg<64, 6, -1>(re, im);
#pragma unroll
    for (int h = 0; h < DH; ++h)
      *(float2*)(pane + (d * DH + h) * ROWB + lane * 8) = make_float2(re[h], im[h]);
  }
  __syncthreads();

  // ---- forward D-FFT: wave <-> h-group, lanes <-> channel ----
  {
    float dre[DD], dim[DD];
    for (int hh = 0; hh < 8; ++hh) {
      const int h = wave * 8 + hh;
#pragma unroll
      for (int d = 0; d < DD; ++d) {
        const float2 v = *(const float2*)(pane + (d * DH + h) * ROWB + lane * 8);
        dre[d] = v.x; dim[d] = v.y;
      }
      fft_reg<16, 4, -1>(dre, dim);
#pragma unroll
      for (int d = 0; d < DD; ++d)
        *(float2*)(pane + (d * DH + h) * ROWB + lane * 8) = make_float2(dre[d], dim[d]);
    }
  }
  __syncthreads();

  // ---- complex block-MLP on WMMA: 64 M-tiles of 16 points, 8 per wave ----
  const int m  = lane & 15;
  const int lh = lane >> 4;

  auto loadB = [&](int layer, int part, int tn) -> v16h {
    const size_t off = ((((size_t)layer * 2 + part) * NB + blk) * 2 + tn) * (32 * 16)
                     + (size_t)lane * 16;
    return *(const v16h*)(wpk + off);
  };
  auto shrink = [](float v) {
    return v > LAMBDA_ ? v - LAMBDA_ : (v < -LAMBDA_ ? v + LAMBDA_ : 0.f);
  };
  const v8f zero = {};

  for (int itt = 0; itt < 8; ++itt) {
    const int rbase = (wave * 8 + itt) * 16;   // 16 pane rows of this tile
    // gather A (re,im) per ISA 16-bit A 16x32 layout; one float2 read yields both
    v16h ar, ai;
#pragma unroll
    for (int i = 0; i < 8; ++i)
#pragma unroll
      for (int j = 0; j < 2; ++j) {
        const int k = ((i < 4) ? 0 : 16) + 8 * lh + 2 * (i & 3) + j;
        const float2 v = *(const float2*)(pane + (rbase + m) * ROWB + k * 8);
        ar[2 * i + j] = (_Float16)v.x;
        ai[2 * i + j] = (_Float16)v.y;
      }
    // layer 1: (ar + i ai)(W1r + i W1i) + b1, ReLU
#pragma unroll
    for (int tn = 0; tn < 2; ++tn) {
      v16h br = loadB(0, 0, tn), bi = loadB(0, 1, tn), nbi;
#pragma unroll
      for (int e = 0; e < 16; ++e) nbi[e] = -bi[e];
      v8f rr = __builtin_amdgcn_wmma_f32_16x16x32_f16(false, ai, false, nbi, (short)0, zero, false, false);
      rr     = __builtin_amdgcn_wmma_f32_16x16x32_f16(false, ar, false, br,  (short)0, rr,   false, false);
      v8f ii = __builtin_amdgcn_wmma_f32_16x16x32_f16(false, ai, false, br,  (short)0, zero, false, false);
      ii     = __builtin_amdgcn_wmma_f32_16x16x32_f16(false, ar, false, bi,  (short)0, ii,   false, false);
#pragma unroll
      for (int r = 0; r < 8; ++r) {            // C/D layout: M = r + 8*lh, N = lane&15
        const int mm = r + 8 * lh;
        const int o  = tn * 16 + (lane & 15);
        float vr = rr[r] + b1[(0 * NB + blk) * BLK + o];
        float vi = ii[r] + b1[(1 * NB + blk) * BLK + o];
        hidr[wave][mm][o] = vr > 0.f ? vr : 0.f;
        hidi[wave][mm][o] = vi > 0.f ? vi : 0.f;
      }
    }
    // layer 2: gather hidden, (hr + i hi)(W2r + i W2i) + b2, softshrink
    v16h a2r, a2i;
#pragma unroll
    for (int i = 0; i < 8; ++i)
#pragma unroll
      for (int j = 0; j < 2; ++j) {
        const int k = ((i < 4) ? 0 : 16) + 8 * lh + 2 * (i & 3) + j;
        a2r[2 * i + j] = (_Float16)hidr[wave][m][k];
        a2i[2 * i + j] = (_Float16)hidi[wave][m][k];
      }
#pragma unroll
    for (int tn = 0; tn < 2; ++tn) {
      v16h br = loadB(1, 0, tn), bi = loadB(1, 1, tn), nbi;
#pragma unroll
      for (int e = 0; e < 16; ++e) nbi[e] = -bi[e];
      v8f rr = __builtin_amdgcn_wmma_f32_16x16x32_f16(false, a2i, false, nbi, (short)0, zero, false, false);
      rr     = __builtin_amdgcn_wmma_f32_16x16x32_f16(false, a2r, false, br,  (short)0, rr,   false, false);
      v8f ii = __builtin_amdgcn_wmma_f32_16x16x32_f16(false, a2i, false, br,  (short)0, zero, false, false);
      ii     = __builtin_amdgcn_wmma_f32_16x16x32_f16(false, a2r, false, bi,  (short)0, ii,   false, false);
#pragma unroll
      for (int r = 0; r < 8; ++r) {
        const int mm = r + 8 * lh;
        const int o  = tn * 16 + (lane & 15);
        const float vr = shrink(rr[r] + b2[(0 * NB + blk) * BLK + o]);
        const float vi = shrink(ii[r] + b2[(1 * NB + blk) * BLK + o]);
        *(float2*)(pane + (rbase + mm) * ROWB + o * 8) = make_float2(vr, vi);
      }
    }
  }
  __syncthreads();

  // ---- inverse D-FFT ----
  {
    float dre[DD], dim[DD];
    for (int hh = 0; hh < 8; ++hh) {
      const int h = wave * 8 + hh;
#pragma unroll
      for (int d = 0; d < DD; ++d) {
        const float2 v = *(const float2*)(pane + (d * DH + h) * ROWB + lane * 8);
        dre[d] = v.x; dim[d] = v.y;
      }
      fft_reg<16, 4, 1>(dre, dim);
#pragma unroll
      for (int d = 0; d < DD; ++d)
        *(float2*)(pane + (d * DH + h) * ROWB + lane * 8) = make_float2(dre[d], dim[d]);
    }
  }
  __syncthreads();

  // ---- inverse H-FFT ----
  for (int rep = 0; rep < 2; ++rep) {
    const int d = wave + 8 * rep;
#pragma unroll
    for (int h = 0; h < DH; ++h) {
      const float2 v = *(const float2*)(pane + (d * DH + h) * ROWB + lane * 8);
      re[h] = v.x; im[h] = v.y;
    }
    fft_reg<64, 6, 1>(re, im);
#pragma unroll
    for (int h = 0; h < DH; ++h)
      *(float2*)(pane + (d * DH + h) * ROWB + lane * 8) = make_float2(re[h], im[h]);
  }
  __syncthreads();

  // ---- write pane back via async LDS->global (ASYNCcnt) ----
  for (int it = 0; it < 64; ++it) {
    const int q   = it * 256 + t;
    const int row = q >> 4;
    const int ch  = q & 15;
    char* ga = (char*)(gpane + (size_t)row * (WF * DC)) + ch * 16;
    async_l2g_b128(ga, (unsigned)(uintptr_t)(pane + row * ROWB + ch * 16));
  }
  wait_async0();   // drain before wave exit (S_ENDPGM also waits-idle)
}

// ---------------- K3: inverse real FFT along W + residual -----------------
__global__ __launch_bounds__(256) void k_fft_w_inv(const float2* __restrict__ Xf,
                                                   const float* __restrict__ x,
                                                   float* __restrict__ out) {
  const int line = blockIdx.x;
  const int c    = threadIdx.x;
  const int h    = line % DH;
  const int bd   = line / DH;
  const float2* src = Xf + (((size_t)bd * DH + h) * WF) * DC + c;
  float re[DW], im[DW];
#pragma unroll
  for (int wf = 0; wf < WF; ++wf) { float2 v = src[(size_t)wf * DC]; re[wf] = v.x; im[wf] = v.y; }
#pragma unroll
  for (int k = WF; k < DW; ++k) { re[k] = re[DW - k]; im[k] = -im[DW - k]; }  // Hermitian extend
  fft_reg<64, 6, 1>(re, im);
  const float  sc    = 1.0f / 256.0f;   // ortho norm for the whole inverse transform
  const float* resid = x   + (((size_t)bd * DH + h) * DW) * DC + c;
  float*       dst   = out + (((size_t)bd * DH + h) * DW) * DC + c;
#pragma unroll
  for (int w = 0; w < DW; ++w) dst[(size_t)w * DC] = re[w] * sc + resid[(size_t)w * DC];
}

// ---------------- host-side launcher --------------------------------------
extern "C" void kernel_launch(void* const* d_in, const int* in_sizes, int n_in,
                              void* d_out, int out_size, void* d_ws, size_t ws_size,
                              hipStream_t stream) {
  (void)in_sizes; (void)n_in; (void)out_size; (void)ws_size;
  const float* x  = (const float*)d_in[0];
  const float* w1 = (const float*)d_in[1];
  const float* b1 = (const float*)d_in[2];
  const float* w2 = (const float*)d_in[3];
  const float* b2 = (const float*)d_in[4];
  float*  out = (float*)d_out;
  float2* Xf  = (float2*)d_ws;
  const size_t xf_bytes = (size_t)DB * DD * DH * WF * DC * sizeof(float2); // 138,412,032 B
  _Float16* wpk = (_Float16*)((char*)d_ws + xf_bytes);

  k_pack_w   <<<128,              256, 0, stream>>>(w1, w2, wpk);
  k_fft_w_fwd<<<DB * DD * DH,     256, 0, stream>>>(x, Xf);
  k_pane     <<<DB * WF * NB,     256, 0, stream>>>(Xf, wpk, b1, b2);
  k_fft_w_inv<<<DB * DD * DH,     256, 0, stream>>>(Xf, x, out);
}
